// PQMatcher1_53120155517070
// MI455X (gfx1250) — matrix operand order repaired and checked by reading, not compiled, not run on hardware
//
#include <hip/hip_runtime.h>
#include <hip/hip_bf16.h>
#include <math.h>

// ---------------------------------------------------------------------------
// PQMatcher on MI455X (gfx1250, wave32, WMMA bf16 + async global->LDS).
//   Parallel phase (all 256 steps at once):
//     k_attn : per (b, t-tile) scores S=4*up·uq - 2||uq||^2, softmax, ctx c512
//     k_gate : c_ = sigmoid(R @ Wg[1024:]^T) * c      (M=16384,N=1024,K=2048)
//     k_gi   : gi = c_ @ W_ih^T + b_ih                (M=16384,N=3072,K=1024)
//   Sequential phase (256 launches, ping-pong v):
//     k_step : gh = v @ W_hh^T (3 gate tiles fused) + GRU elementwise update
// GEMMs use v_wmma_f32_16x16x32_bf16 with 4x4 register blocking per wave.
// k_gate/k_gi stage the (workgroup-shared) B tile in LDS once per K-step via
// global_load_async_to_lds_b128 (double buffered, s_wait_asynccnt), removing
// the 4x redundant per-wave weight traffic from L2.
// ---------------------------------------------------------------------------

typedef __bf16 bf16_t;
typedef __attribute__((ext_vector_type(16))) __bf16 v16bf;
typedef __attribute__((ext_vector_type(8)))  float  v8f;

#define LPc  256
#define LQc  256
#define Bc   64
#define INc  512
#define IN2c 1024
#define Hc   1024
#define Mc   (LPc * Bc)      // 16384 flattened (t,b) rows

// ---- WMMA fragment loaders (layouts per CDNA5 ISA 7.12.2) -------

// A: 16x32 bf16, row-major source, leading dim lda.
// lane<16: row=lane, K={0..7, 16..23}; lane>=16: row=lane-16, K={8..15, 24..31}
__device__ __forceinline__ v16bf load_a_frag(const bf16_t* A, int lda, int lane) {
  const int row = lane & 15;
  const int kb  = (lane >> 4) * 8;
  const bf16_t* p = A + (long)row * lda;
  v16bf f;
#pragma unroll
  for (int i = 0; i < 8; ++i) f[i] = p[kb + i];
#pragma unroll
  for (int i = 0; i < 8; ++i) f[8 + i] = p[16 + kb + i];
  return f;
}

// B: 32x16 bf16 where B[k,n] = W[n,k] (weight stored N-major, leading dim ldw).
// lane<16: col=lane, K={0..7, 16..23}; lane>=16: col=lane-16, K={8..15, 24..31}
__device__ __forceinline__ v16bf load_b_frag_nk(const bf16_t* W, int ldw, int lane) {
  const int col = lane & 15;
  const int kb  = (lane >> 4) * 8;
  const bf16_t* p = W + (long)col * ldw + kb;
  v16bf f;
#pragma unroll
  for (int i = 0; i < 8; ++i) f[i] = p[i];
#pragma unroll
  for (int i = 0; i < 8; ++i) f[8 + i] = p[16 + i];
  return f;
}

__device__ __forceinline__ v8f wmma_bf16(v16bf a, v16bf b, v8f c) {
  return __builtin_amdgcn_wmma_f32_16x16x32_bf16(false, a, false, b, (short)0, c,
                                                 false, false);
}

__device__ __forceinline__ float sigmoidf_(float x) {
  return 1.0f / (1.0f + __expf(-x));
}

// ---- async global -> LDS staging (gfx1250, ASYNCcnt-tracked) ----------------
// Per-lane 16B copy; LDS address = low 32 bits of generic pointer (ISA 10.2:
// flat->LDS discards upper bits).
__device__ __forceinline__ void async_cp16(void* lds, const void* gmem) {
  asm volatile("global_load_async_to_lds_b128 %0, %1, off"
               :: "v"((unsigned int)(unsigned long long)lds),
                  "v"((unsigned long long)gmem)
               : "memory");
}
#define WAIT_ASYNC(N) asm volatile("s_wait_asynccnt " #N ::: "memory")

// stage a 64-row x 32-col bf16 tile (4KB) of W[n0.., k0..] into LDS, 128 thr
__device__ __forceinline__ void stage_b64x32(bf16_t (*dst)[32],
                                             const bf16_t* __restrict__ W,
                                             long ldw, int n0, int k0, int tid) {
#pragma unroll
  for (int c = 0; c < 2; ++c) {
    const int e  = tid + c * 128;        // 16B chunk id, 0..255
    const int r  = e >> 2;               // row 0..63
    const int kk = (e & 3) * 8;          // col offset (elements)
    async_cp16(&dst[r][kk], W + (long)(n0 + r) * ldw + k0 + kk);
  }
}

// ---- setup kernels ----------------------------------------------------------

__global__ void k_cvt_bf16(bf16_t* __restrict__ dst, const float* __restrict__ src,
                           int n) {
  int i = blockIdx.x * blockDim.x + threadIdx.x;
  if (i < n) dst[i] = (bf16_t)src[i];
}

// ||uq[q,b,:]||^2 over 512 dims, one thread per (q*B+b)
__global__ void k_uqn(float* __restrict__ uqn, const float* __restrict__ uq) {
  int i = blockIdx.x * blockDim.x + threadIdx.x;
  if (i >= LQc * Bc) return;
  const float* p = uq + (long)i * INc;
  float s = 0.f;
#pragma unroll 4
  for (int k = 0; k < INc; ++k) { float v = p[k]; s += v * v; }
  uqn[i] = s;
}

// transpose uq -> uqT[b][n][q] (feature-major) for contiguous B-frag loads
__global__ void k_tq(bf16_t* __restrict__ uqT, const float* __restrict__ uq) {
  long i = (long)blockIdx.x * blockDim.x + threadIdx.x;   // i = (b*IN + n)*LQ + q
  if (i >= (long)Bc * INc * LQc) return;
  int q = (int)(i & (LQc - 1));
  long bn = i >> 8;                 // b*IN + n
  int n = (int)(bn & (INc - 1));
  int b = (int)(bn >> 9);
  uqT[i] = (bf16_t)uq[((long)q * Bc + b) * INc + n];
}

__global__ void k_vinit(float* __restrict__ vf, bf16_t* __restrict__ vb,
                        const float* __restrict__ v0) {
  int i = blockIdx.x * blockDim.x + threadIdx.x;
  if (i < Bc * Hc) { float v = v0[i]; vf[i] = v; vb[i] = (bf16_t)v; }
}

// ---- attention: per (t-tile, b): S -> softmax -> c512 -----------------------
// grid (LP/16, B), block 128 (4 waves). upb/uqb natural layout [t or q][b][k].
__global__ void k_attn(bf16_t* __restrict__ c512b, const bf16_t* __restrict__ upb,
                       const bf16_t* __restrict__ uqb, const bf16_t* __restrict__ uqT,
                       const float* __restrict__ uqn) {
  __shared__ float  S[16][LQc];
  __shared__ bf16_t Asm[16][LQc];
  const int tid  = threadIdx.x;
  const int lane = tid & 31;
  const int wave = tid >> 5;
  const int t0   = blockIdx.x * 16;
  const int b    = blockIdx.y;
  const int ldu  = Bc * INc;                 // row stride (t or q) in bf16 elems
  const bf16_t* Ab = upb + (long)b * INc;    // Up_b[t,k]
  const bf16_t* Qb = uqb + (long)b * INc;    // Uq_b[q,k]
  const bf16_t* QT = uqT + (long)b * INc * LQc;  // UqT_b[n,q], ld = LQ

  // phase 1: S tile [16 x 256] = Up_b[t0:t0+16] @ Uq_b^T, K=512
  // wave handles q-tiles [wave*4, wave*4+4): A-frag loaded once per K-step.
  {
    v8f acc[4] = {};
#pragma unroll 2
    for (int kt = 0; kt < INc / 32; ++kt) {
      v16bf a = load_a_frag(Ab + (long)t0 * ldu + kt * 32, ldu, lane);
#pragma unroll
      for (int j = 0; j < 4; ++j) {
        v16bf bf = load_b_frag_nk(Qb + (long)((wave * 4 + j) * 16) * ldu + kt * 32,
                                  ldu, lane);
        acc[j] = wmma_bf16(a, bf, acc[j]);
      }
    }
    const int col = lane & 15, rb = (lane >> 4) * 8;
#pragma unroll
    for (int j = 0; j < 4; ++j) {
      const int q = (wave * 4 + j) * 16 + col;
      const float nq = uqn[q * Bc + b];
#pragma unroll
      for (int i = 0; i < 8; ++i) S[rb + i][q] = 4.0f * acc[j][i] - 2.0f * nq;
    }
  }
  __syncthreads();

  // softmax over q: one thread per row
  if (tid < 16) {
    float mx = -3.0e38f;
    for (int q = 0; q < LQc; ++q) mx = fmaxf(mx, S[tid][q]);
    float sum = 0.f;
    for (int q = 0; q < LQc; ++q) { float e = __expf(S[tid][q] - mx); sum += e; S[tid][q] = e; }
    float inv = 1.0f / sum;
    for (int q = 0; q < LQc; ++q) Asm[tid][q] = (bf16_t)(S[tid][q] * inv);
  }
  __syncthreads();

  // phase 2: C tile [16 x 512] = A_soft @ Uq_b, K=256.
  // wave handles n-tiles [wave*8, wave*8+8); B from transposed uqT (contiguous).
  {
    v8f acc[8] = {};
#pragma unroll
    for (int kt = 0; kt < LQc / 32; ++kt) {
      v16bf a = load_a_frag(&Asm[0][0] + kt * 32, LQc, lane);
#pragma unroll
      for (int j = 0; j < 8; ++j) {
        v16bf bf = load_b_frag_nk(QT + (long)((wave * 8 + j) * 16) * LQc + kt * 32,
                                  LQc, lane);
        acc[j] = wmma_bf16(a, bf, acc[j]);
      }
    }
    const int col = lane & 15, rb = (lane >> 4) * 8;
#pragma unroll
    for (int j = 0; j < 8; ++j) {
#pragma unroll
      for (int i = 0; i < 8; ++i) {
        long m = (long)(t0 + rb + i) * Bc + b;       // flattened (t,b) row
        c512b[m * INc + (wave * 8 + j) * 16 + col] = (bf16_t)acc[j][i];
      }
    }
  }
}

// ---- gating: c_ = sigmoid(R @ Wg2^T) * c ------------------------------------
// R[m] = [up_m, up_m, c_m, c_m] (virtual, K=2048). Wg2 = Wg rows 1024..2047.
// grid (N/64=16, M/256=64), block 128: wave w owns an independent 64(M)x64(N)
// block; the workgroup-shared 64x32 B tile is async-staged in LDS (2 buffers).
__global__ void k_gate(bf16_t* __restrict__ cb, const bf16_t* __restrict__ upb,
                       const bf16_t* __restrict__ c512b,
                       const bf16_t* __restrict__ wg2b) {
  __shared__ bf16_t Bsm[2][64][32];
  const int tid = threadIdx.x;
  const int lane = tid & 31, wave = tid >> 5;
  const int n0 = blockIdx.x * 64;
  const int m0 = (blockIdx.y * 4 + wave) * 64;
  const int NK = (2 * IN2c) / 32;
  v8f acc[4][4] = {};
  stage_b64x32(Bsm[0], wg2b, 2 * IN2c, n0, 0, tid);
  for (int kt = 0; kt < NK; ++kt) {
    const int buf = kt & 1;
    if (kt + 1 < NK) {
      stage_b64x32(Bsm[buf ^ 1], wg2b, 2 * IN2c, n0, (kt + 1) * 32, tid);
      WAIT_ASYNC(0x2);                 // current buffer's 2 per-wave ops done
    } else {
      WAIT_ASYNC(0x0);
    }
    __syncthreads();                   // publish staged tile to all waves
    const int k0 = kt * 32;
    const bf16_t* Abase = (k0 < IN2c) ? upb : c512b;   // halves repeat 512-dim part
    const int kk = k0 & (INc - 1);
    v16bf a[4], bf[4];
#pragma unroll
    for (int i = 0; i < 4; ++i)
      a[i] = load_a_frag(Abase + (long)(m0 + i * 16) * INc + kk, INc, lane);
#pragma unroll
    for (int j = 0; j < 4; ++j)
      bf[j] = load_b_frag_nk(&Bsm[buf][j * 16][0], 32, lane);
#pragma unroll
    for (int i = 0; i < 4; ++i)
#pragma unroll
      for (int j = 0; j < 4; ++j) acc[i][j] = wmma_bf16(a[i], bf[j], acc[i][j]);
    __syncthreads();                   // all reads done before buf is re-staged
  }
  const int col = lane & 15, rb = (lane >> 4) * 8;
#pragma unroll
  for (int i = 0; i < 4; ++i) {
#pragma unroll
    for (int j = 0; j < 4; ++j) {
      const int n = n0 + j * 16 + col, nc = n & (INc - 1);
#pragma unroll
      for (int r = 0; r < 8; ++r) {
        long m = m0 + i * 16 + rb + r;
        float g = sigmoidf_(acc[i][j][r]);
        float cv = (float)c512b[m * INc + nc];
        cb[m * IN2c + n] = (bf16_t)(g * cv);
      }
    }
  }
}

// ---- gi = c_ @ W_ih^T + b_ih -> bf16 [16384, 3072] --------------------------
// grid (3H/64=48, M/256=64), block 128; same async-LDS B staging as k_gate.
__global__ void k_gi(bf16_t* __restrict__ gib, const bf16_t* __restrict__ cb,
                     const bf16_t* __restrict__ wihb, const float* __restrict__ bih) {
  __shared__ bf16_t Bsm[2][64][32];
  const int tid = threadIdx.x;
  const int lane = tid & 31, wave = tid >> 5;
  const int n0 = blockIdx.x * 64;
  const int m0 = (blockIdx.y * 4 + wave) * 64;
  const int NK = IN2c / 32;
  v8f acc[4][4] = {};
  stage_b64x32(Bsm[0], wihb, IN2c, n0, 0, tid);
  for (int kt = 0; kt < NK; ++kt) {
    const int buf = kt & 1;
    if (kt + 1 < NK) {
      stage_b64x32(Bsm[buf ^ 1], wihb, IN2c, n0, (kt + 1) * 32, tid);
      WAIT_ASYNC(0x2);
    } else {
      WAIT_ASYNC(0x0);
    }
    __syncthreads();
    v16bf a[4], bf[4];
#pragma unroll
    for (int i = 0; i < 4; ++i)
      a[i] = load_a_frag(cb + (long)(m0 + i * 16) * IN2c + kt * 32, IN2c, lane);
#pragma unroll
    for (int j = 0; j < 4; ++j)
      bf[j] = load_b_frag_nk(&Bsm[buf][j * 16][0], 32, lane);
#pragma unroll
    for (int i = 0; i < 4; ++i)
#pragma unroll
      for (int j = 0; j < 4; ++j) acc[i][j] = wmma_bf16(a[i], bf[j], acc[i][j]);
    __syncthreads();
  }
  const int col = lane & 15, rb = (lane >> 4) * 8;
#pragma unroll
  for (int j = 0; j < 4; ++j) {
    const int n = n0 + j * 16 + col;
    const float bias = bih[n];
#pragma unroll
    for (int i = 0; i < 4; ++i)
#pragma unroll
      for (int r = 0; r < 8; ++r)
        gib[(long)(m0 + i * 16 + rb + r) * (3 * Hc) + n] = (bf16_t)(acc[i][j][r] + bias);
  }
}

// ---- sequential step: gh (3 gate tiles) + GRU elementwise -------------------
// grid 64 (h-tiles), block 128 (wave w = b-tile w). Direct loads (L2-resident;
// this is the latency-critical chain, no per-iteration barriers). Ping-pong v.
__global__ void k_step(float* __restrict__ out, const bf16_t* __restrict__ gib,
                       const bf16_t* __restrict__ whhb, const float* __restrict__ bhh,
                       const float* __restrict__ vf_in, const bf16_t* __restrict__ vb_in,
                       float* __restrict__ vf_out, bf16_t* __restrict__ vb_out,
                       int t) {
  const int lane = threadIdx.x & 31, wave = threadIdx.x >> 5;
  const int h0 = blockIdx.x * 16;
  const int b0 = wave * 16;
  v8f ar = {}, az = {}, an = {};
#pragma unroll 2
  for (int kt = 0; kt < Hc / 32; ++kt) {
    v16bf a  = load_a_frag(vb_in + (long)b0 * Hc + kt * 32, Hc, lane);
    v16bf br = load_b_frag_nk(whhb + (long)h0 * Hc + kt * 32, Hc, lane);
    v16bf bz = load_b_frag_nk(whhb + (long)(Hc + h0) * Hc + kt * 32, Hc, lane);
    v16bf bn = load_b_frag_nk(whhb + (long)(2 * Hc + h0) * Hc + kt * 32, Hc, lane);
    ar = wmma_bf16(a, br, ar);
    az = wmma_bf16(a, bz, az);
    an = wmma_bf16(a, bn, an);
  }
  const int col = lane & 15, rb = (lane >> 4) * 8;
  const int h = h0 + col;
  const float bhr = bhh[h], bhz = bhh[Hc + h], bhn = bhh[2 * Hc + h];
#pragma unroll
  for (int i = 0; i < 8; ++i) {
    const int b = b0 + rb + i;
    const long mg = (long)(t * Bc + b) * (3 * Hc);
    float gr = (float)gib[mg + h]          + ar[i] + bhr;
    float gz = (float)gib[mg + Hc + h]     + az[i] + bhz;
    float gn = (float)gib[mg + 2 * Hc + h];
    float rr = sigmoidf_(gr);
    float zz = sigmoidf_(gz);
    float nn = tanhf(gn + rr * (an[i] + bhn));
    float vp = vf_in[(long)b * Hc + h];
    float vnew = (1.0f - zz) * nn + zz * vp;
    out[(long)(t * Bc + b) * Hc + h] = vnew;
    vf_out[(long)b * Hc + h] = vnew;
    vb_out[(long)b * Hc + h] = (bf16_t)vnew;
  }
}

// ---------------------------------------------------------------------------

extern "C" void kernel_launch(void* const* d_in, const int* in_sizes, int n_in,
                              void* d_out, int out_size, void* d_ws, size_t ws_size,
                              hipStream_t stream) {
  (void)in_sizes; (void)n_in; (void)out_size; (void)ws_size;
  const float* up  = (const float*)d_in[0];
  const float* uq  = (const float*)d_in[1];
  const float* v0  = (const float*)d_in[2];
  const float* Wg  = (const float*)d_in[3];
  const float* Wih = (const float*)d_in[4];
  const float* Whh = (const float*)d_in[5];
  const float* bih = (const float*)d_in[6];
  const float* bhh = (const float*)d_in[7];
  float* out = (float*)d_out;

  // workspace carve-up (~220 MB total)
  char* ws = (char*)d_ws;
  size_t off = 0;
  auto carve = [&](size_t bytes) -> void* {
    void* p = ws + off;
    off = (off + bytes + 255) & ~(size_t)255;
    return p;
  };
  bf16_t* upb   = (bf16_t*)carve((size_t)Mc * INc * 2);
  bf16_t* uqb   = (bf16_t*)carve((size_t)LQc * Bc * INc * 2);
  bf16_t* uqT   = (bf16_t*)carve((size_t)Bc * INc * LQc * 2);
  float*  uqn   = (float*) carve((size_t)LQc * Bc * 4);
  bf16_t* c512b = (bf16_t*)carve((size_t)Mc * INc * 2);
  bf16_t* wg2b  = (bf16_t*)carve((size_t)IN2c * 2 * IN2c * 2);
  bf16_t* wihb  = (bf16_t*)carve((size_t)3 * Hc * IN2c * 2);
  bf16_t* whhb  = (bf16_t*)carve((size_t)3 * Hc * Hc * 2);
  bf16_t* cb    = (bf16_t*)carve((size_t)Mc * IN2c * 2);
  bf16_t* gib   = (bf16_t*)carve((size_t)Mc * 3 * Hc * 2);
  float*  vf0   = (float*) carve((size_t)Bc * Hc * 4);
  float*  vf1   = (float*) carve((size_t)Bc * Hc * 4);
  bf16_t* vb0   = (bf16_t*)carve((size_t)Bc * Hc * 2);
  bf16_t* vb1   = (bf16_t*)carve((size_t)Bc * Hc * 2);

  const int T = 256;
  // bf16 conversions (Wg: only rows 1024..2047 needed -> contiguous 2nd half)
  k_cvt_bf16<<<(Mc * INc + T - 1) / T, T, 0, stream>>>(upb, up, Mc * INc);
  k_cvt_bf16<<<(LQc * Bc * INc + T - 1) / T, T, 0, stream>>>(uqb, uq, LQc * Bc * INc);
  k_tq<<<(Bc * INc * LQc + T - 1) / T, T, 0, stream>>>(uqT, uq);
  k_cvt_bf16<<<(IN2c * 2 * IN2c + T - 1) / T, T, 0, stream>>>(
      wg2b, Wg + (size_t)IN2c * 2 * IN2c, IN2c * 2 * IN2c);
  k_cvt_bf16<<<(3 * Hc * IN2c + T - 1) / T, T, 0, stream>>>(wihb, Wih, 3 * Hc * IN2c);
  k_cvt_bf16<<<(3 * Hc * Hc + T - 1) / T, T, 0, stream>>>(whhb, Whh, 3 * Hc * Hc);
  k_uqn<<<(LQc * Bc + T - 1) / T, T, 0, stream>>>(uqn, uq);
  k_vinit<<<(Bc * Hc + T - 1) / T, T, 0, stream>>>(vf0, vb0, v0);

  // parallel phase
  k_attn<<<dim3(LPc / 16, Bc), 128, 0, stream>>>(c512b, upb, uqb, uqT, uqn);
  k_gate<<<dim3(IN2c / 64, Mc / 256), 128, 0, stream>>>(cb, upb, c512b, wg2b);
  k_gi<<<dim3(3 * Hc / 64, Mc / 256), 128, 0, stream>>>(gib, cb, wihb, bih);

  // sequential GRU scan, ping-pong v
  float*  vf[2] = {vf0, vf1};
  bf16_t* vb[2] = {vb0, vb1};
  for (int t = 0; t < LPc; ++t) {
    k_step<<<Hc / 16, 128, 0, stream>>>(out, gib, whhb, bhh,
                                        vf[t & 1], vb[t & 1],
                                        vf[(t + 1) & 1], vb[(t + 1) & 1], t);
  }
}
